// RGCNEncoder_67456756351571
// MI455X (gfx1250) — matrix operand order repaired and checked by reading, compiled, not run on hardware
//
#include <hip/hip_runtime.h>
#include <hip/hip_bf16.h>
#include <stdint.h>

// ---------------------------------------------------------------------------
// RGCN encoder for MI455X (gfx1250, wave32, WMMA, async LDS fill).
//
//   x = fallback; x[file_idx]=file@Wf+bf; x[dom_idx]=dom@Wd+bd; x[ip_idx]=ip@Wi+bi
//   per layer: segment-mean over (dst,rel) -> [N,R,128]; fused GEMM
//     [N, R*128+128] @ [R*128+128, 128]  (tail 128 K-rows = root, i.e. x@root)
//     + bias; LN (+ReLU after layer 1).
//
// GEMM: v_wmma_f32_16x16x32_bf16. B pre-packed in the WMMA pair layout so each
// lane's B fragment is two contiguous global_load_b128; A tile staged in LDS
// via global_load_async_to_lds_b128 (ASYNCcnt) and read as two ds_load_b128.
// Edge aggregation uses hardware global_atomic_add_f32.
// ---------------------------------------------------------------------------

typedef __attribute__((ext_vector_type(16))) __bf16 v16bf;
typedef __attribute__((ext_vector_type(8)))  float  v8f;

#define HID   128
#define NREL  8
#define KEXT  (NREL * HID + HID)   // 1152

union frag_u {
    uint32_t u[8];
    uint4    q[2];
    v16bf    v;
};

union pair_u {
    __bf16   h[2];
    uint32_t u;
};

// ---------------------------------------------------------------------------
// Elementwise helpers (b128 vectorized)
// ---------------------------------------------------------------------------
__global__ __launch_bounds__(256) void zero_v4_kernel(float4* __restrict__ p, int n4) {
    int i = blockIdx.x * 256 + threadIdx.x;
    if (i < n4) p[i] = make_float4(0.f, 0.f, 0.f, 0.f);
}

__global__ __launch_bounds__(256) void copy_v4_kernel(const float4* __restrict__ s,
                                                      float4* __restrict__ d, int n4) {
    int i = blockIdx.x * 256 + threadIdx.x;
    if (i < n4) d[i] = s[i];
}

// f32 -> bf16, two elements per thread (dword stores)
__global__ __launch_bounds__(256) void cvt_pack2_kernel(const float* __restrict__ src,
                                                        uint32_t* __restrict__ dst, int n2) {
    int i = blockIdx.x * 256 + threadIdx.x;
    if (i >= n2) return;
    pair_u p;
    p.h[0] = (__bf16)src[2 * i];
    p.h[1] = (__bf16)src[2 * i + 1];
    dst[i] = p.u;
}

// ---------------------------------------------------------------------------
// Edge scatter: one wave per edge; lane owns 4 consecutive feats (b128 gather,
// 4 hardware f32 atomics). cnt[dst*R+rel] += 1.
// ---------------------------------------------------------------------------
__global__ __launch_bounds__(256)
void scatter_mean_kernel(const int* __restrict__ edge_index,   // [2,E]
                         const int* __restrict__ edge_type,    // [E]
                         const float* __restrict__ x,          // [N,128]
                         float* __restrict__ msum,             // [N*R,128]
                         float* __restrict__ cnt,              // [N*R]
                         int E) {
    int e    = blockIdx.x * 8 + (threadIdx.x >> 5);
    int lane = threadIdx.x & 31;
    if (e >= E) return;
    int src = edge_index[e];
    int dst = edge_index[E + e];
    int rel = edge_type[e];
    int seg = dst * NREL + rel;
    float4 v = ((const float4*)(x + (size_t)src * HID))[lane];
    float* ms = msum + (size_t)seg * HID + lane * 4;
    unsafeAtomicAdd(&ms[0], v.x);
    unsafeAtomicAdd(&ms[1], v.y);
    unsafeAtomicAdd(&ms[2], v.z);
    unsafeAtomicAdd(&ms[3], v.w);
    if (lane == 0) unsafeAtomicAdd(&cnt[seg], 1.0f);
}

// ---------------------------------------------------------------------------
// Pack B matrices into the WMMA-B pair layout:
//   Bpack[kb][col][d] (dword) = { B[kb*32+2d][col], B[kb*32+2d+1][col] }
// so lane (half,n) of wave w reads 8 contiguous dwords at
//   (((kb*128)+col)*16 + half*8).
// ---------------------------------------------------------------------------
__global__ __launch_bounds__(256)
void pack_b_kernel(const float* __restrict__ src,   // [K,128] row-major f32
                   uint32_t* __restrict__ dst, int K) {
    int idx = blockIdx.x * 256 + threadIdx.x;
    int total = (K / 32) * 128 * 16;
    if (idx >= total) return;
    int d   = idx & 15;
    int col = (idx >> 4) & 127;
    int kb  = idx >> 11;
    int k   = kb * 32 + 2 * d;
    pair_u p;
    p.h[0] = (__bf16)src[(size_t)k * HID + col];
    p.h[1] = (__bf16)src[(size_t)(k + 1) * HID + col];
    dst[idx] = p.u;
}

// Packed B for the RGCN layer GEMM: rows 0..1023 = comp.bases, 1024..1151 = root.
__global__ __launch_bounds__(256)
void pack_rgcn_b_kernel(const float* __restrict__ comp,    // [R,4]
                        const float* __restrict__ bases,   // [4,128,128]
                        const float* __restrict__ root,    // [128,128]
                        uint32_t* __restrict__ dst) {      // [(KEXT/32)*128*16]
    int idx = blockIdx.x * 256 + threadIdx.x;
    if (idx >= (KEXT / 32) * 128 * 16) return;
    int d   = idx & 15;
    int col = (idx >> 4) & 127;
    int kb  = idx >> 11;
    int k0  = kb * 32 + 2 * d;
    pair_u p;
#pragma unroll
    for (int t = 0; t < 2; ++t) {
        int k = k0 + t;
        float acc;
        if (k < NREL * HID) {
            int r = k >> 7;
            int i = k & 127;
            acc = 0.0f;
#pragma unroll
            for (int b = 0; b < 4; ++b)
                acc += comp[r * 4 + b] * bases[((b * HID) + i) * HID + col];
        } else {
            acc = root[(k - NREL * HID) * HID + col];
        }
        p.h[t] = (__bf16)acc;
    }
    dst[idx] = p.u;
}

// ---------------------------------------------------------------------------
// mean = msum / max(cnt,1) packed to bf16 A [N,1152]; tail 128 cols = x.
// One dword (2 bf16) per thread.
// ---------------------------------------------------------------------------
__global__ __launch_bounds__(256)
void meanpack_kernel(const float* __restrict__ msum,  // [N*R,128]
                     const float* __restrict__ cnt,   // [N*R]
                     const float* __restrict__ x,     // [N,128]
                     uint32_t* __restrict__ Apack32,  // [N,KEXT/2] dwords
                     int N) {
    int idx = blockIdx.x * 256 + threadIdx.x;
    if (idx >= N * (KEXT / 2)) return;
    int n  = idx / (KEXT / 2);
    int c  = (idx - n * (KEXT / 2)) * 2;
    float v0, v1;
    if (c < NREL * HID) {
        int seg = n * NREL + (c >> 7);
        float inv = 1.0f / fmaxf(cnt[seg], 1.0f);
        const float* mp = msum + (size_t)seg * HID + (c & 127);
        v0 = mp[0] * inv;
        v1 = mp[1] * inv;
    } else {
        const float* xp = x + (size_t)n * HID + (c - NREL * HID);
        v0 = xp[0];
        v1 = xp[1];
    }
    pair_u p;
    p.h[0] = (__bf16)v0;
    p.h[1] = (__bf16)v1;
    Apack32[idx] = p.u;
}

// ---------------------------------------------------------------------------
// WMMA GEMM:  C[M,128] = A[M,K](bf16) @ B[K,128](bf16, pre-packed) + bias.
// 8 waves per block; block owns 16 rows; wave w owns columns [16w,16w+16).
// A tile (16x32 bf16 = 1 KB) filled with global_load_async_to_lds_b128;
// fragments are 2x ds_load_b128 (A) + 2x global_load_b128 (packed B).
// ---------------------------------------------------------------------------
template <bool HAS_ROWMAP>
__global__ __launch_bounds__(256)
void wmma_gemm128_kernel(const uint32_t* __restrict__ A32, int lda,   // lda in bf16 elems
                         const uint32_t* __restrict__ Bp32,           // packed B
                         const float* __restrict__ bias,              // [128]
                         float* __restrict__ C,                       // [*,128]
                         const int* __restrict__ rowmap,
                         int M, int K) {
    __shared__ uint32_t Alds32[16 * 16];   // 16 rows x 16 dwords (32 bf16)

    const int tid  = threadIdx.x;
    const int wave = tid >> 5;
    const int lane = tid & 31;
    const int half = lane >> 4;
    const int m    = lane & 15;
    const int col  = wave * 16 + (lane & 15);
    const int m0   = blockIdx.x * 16;

    v8f acc = {};
    const int nkb = K >> 5;

    for (int kb = 0; kb < nkb; ++kb) {
        // ---- async stage A tile: 64 lanes x b128 = 16 rows x 64 bytes
        if (tid < 64) {
            int r   = tid >> 2;
            int c4  = (tid & 3) * 4;                       // dword within row
            int row = m0 + r;
            if (row >= M) row = M - 1;                     // clamp; stores guarded
            const uint32_t* gp = A32 + (((size_t)row * lda + kb * 32) >> 1) + c4;
            uint32_t lds_off = (uint32_t)(uintptr_t)(&Alds32[r * 16 + c4]);
            asm volatile("global_load_async_to_lds_b128 %0, %1, off"
                         :: "v"(lds_off), "v"((uint64_t)(uintptr_t)gp)
                         : "memory");
        }
        asm volatile("s_wait_asynccnt 0x0" ::: "memory");
        __syncthreads();

        // ---- A fragment: two contiguous 16B LDS reads per lane
        frag_u fa;
        fa.q[0] = *(const uint4*)(&Alds32[m * 16 + half * 8]);
        fa.q[1] = *(const uint4*)(&Alds32[m * 16 + half * 8 + 4]);

        // ---- B fragment: two contiguous 16B global reads (pre-packed layout)
        const uint32_t* bp = Bp32 + (((size_t)kb * HID + col) << 4) + half * 8;
        frag_u fb;
        fb.q[0] = *(const uint4*)(bp);
        fb.q[1] = *(const uint4*)(bp + 4);

        acc = __builtin_amdgcn_wmma_f32_16x16x32_bf16(
            false, fa.v, false, fb.v, (short)0, acc, false, false);

        __syncthreads();
    }

    // ---- store: VGPR i + 8*half = row, lane n = column
    const float bv = bias[col];
#pragma unroll
    for (int i = 0; i < 8; ++i) {
        int row = m0 + half * 8 + i;
        if (row < M) {
            int drow = HAS_ROWMAP ? rowmap[row] : row;
            C[(size_t)drow * HID + col] = acc[i] + bv;
        }
    }
}

// ---------------------------------------------------------------------------
// LayerNorm over 128 feats, one wave per row, b128 loads/stores, wave32 shfl.
// ---------------------------------------------------------------------------
__global__ __launch_bounds__(256)
void layernorm_kernel(const float* __restrict__ in, const float* __restrict__ g,
                      const float* __restrict__ b, float* __restrict__ out,
                      int N, int relu) {
    int row  = blockIdx.x * 8 + (threadIdx.x >> 5);
    int lane = threadIdx.x & 31;
    if (row >= N) return;
    float4 v = ((const float4*)(in + (size_t)row * HID))[lane];
    float s = v.x + v.y + v.z + v.w;
#pragma unroll
    for (int msk = 16; msk >= 1; msk >>= 1) s += __shfl_xor(s, msk, 32);
    float mu = s * (1.0f / 128.0f);
    float dx = v.x - mu, dy = v.y - mu, dz = v.z - mu, dw = v.w - mu;
    float q = dx * dx + dy * dy + dz * dz + dw * dw;
#pragma unroll
    for (int msk = 16; msk >= 1; msk >>= 1) q += __shfl_xor(q, msk, 32);
    float inv = rsqrtf(q * (1.0f / 128.0f) + 1e-5f);
    float4 gg = ((const float4*)g)[lane];
    float4 bb = ((const float4*)b)[lane];
    float4 y;
    y.x = dx * inv * gg.x + bb.x;
    y.y = dy * inv * gg.y + bb.y;
    y.z = dz * inv * gg.z + bb.z;
    y.w = dw * inv * gg.w + bb.w;
    if (relu) {
        y.x = fmaxf(y.x, 0.f); y.y = fmaxf(y.y, 0.f);
        y.z = fmaxf(y.z, 0.f); y.w = fmaxf(y.w, 0.f);
    }
    ((float4*)(out + (size_t)row * HID))[lane] = y;
}

// ---------------------------------------------------------------------------
// Host orchestration
// ---------------------------------------------------------------------------
extern "C" void kernel_launch(void* const* d_in, const int* in_sizes, int n_in,
                              void* d_out, int out_size, void* d_ws, size_t ws_size,
                              hipStream_t stream) {
    (void)n_in; (void)out_size; (void)ws_size;
    const int N = 50000;
    const int E = in_sizes[0] / 2;

    const int*   edge_index = (const int*)d_in[0];
    const int*   edge_type  = (const int*)d_in[1];
    const float* file_feats = (const float*)d_in[2];
    const int*   file_idx   = (const int*)d_in[3];
    const float* dom_feats  = (const float*)d_in[4];
    const int*   dom_idx    = (const int*)d_in[5];
    const float* ip_feats   = (const float*)d_in[6];
    const int*   ip_idx     = (const int*)d_in[7];
    const float* fallback   = (const float*)d_in[8];
    const float* Wf = (const float*)d_in[9];   const float* bfv = (const float*)d_in[10];
    const float* Wd = (const float*)d_in[11];  const float* bdv = (const float*)d_in[12];
    const float* Wi = (const float*)d_in[13];  const float* biv = (const float*)d_in[14];
    const float* comp1  = (const float*)d_in[15];
    const float* bases1 = (const float*)d_in[16];
    const float* root1  = (const float*)d_in[17];
    const float* bias1  = (const float*)d_in[18];
    const float* comp2  = (const float*)d_in[19];
    const float* bases2 = (const float*)d_in[20];
    const float* root2  = (const float*)d_in[21];
    const float* bias2  = (const float*)d_in[22];
    const float* ln1g = (const float*)d_in[23];
    const float* ln1b = (const float*)d_in[24];
    const float* ln2g = (const float*)d_in[25];
    const float* ln2b = (const float*)d_in[26];

    // Workspace partition
    char*  ws  = (char*)d_ws;
    size_t off = 0;
    auto alloc = [&](size_t bytes) -> void* {
        void* p = ws + off;
        off = (off + bytes + 255) & ~(size_t)255;
        return p;
    };
    float*    xbuf  = (float*)alloc((size_t)N * HID * 4);
    float*    hbuf  = (float*)alloc((size_t)N * HID * 4);
    float*    msum  = (float*)alloc((size_t)N * NREL * HID * 4);
    float*    cnt   = (float*)alloc((size_t)N * NREL * 4);
    uint32_t* Apack = (uint32_t*)alloc((size_t)N * (KEXT / 2) * 4);
    uint32_t* Bpack = (uint32_t*)alloc((size_t)(KEXT / 32) * 128 * 16 * 4);
    uint32_t* featb = (uint32_t*)alloc((size_t)20000 * 128 / 2 * 4);

    const int NH = N * HID;

    // ---- build_x: fallback then per-type projections (rowmap-scattered)
    copy_v4_kernel<<<(NH / 4 + 255) / 256, 256, 0, stream>>>(
        (const float4*)fallback, (float4*)xbuf, NH / 4);

    // file: [20000,128] @ [128,128]
    cvt_pack2_kernel<<<(20000 * 128 / 2 + 255) / 256, 256, 0, stream>>>(file_feats, featb, 20000 * 128 / 2);
    pack_b_kernel<<<((128 / 32) * 128 * 16 + 255) / 256, 256, 0, stream>>>(Wf, Bpack, 128);
    wmma_gemm128_kernel<true><<<(20000 + 15) / 16, 256, 0, stream>>>(
        featb, 128, Bpack, bfv, xbuf, file_idx, 20000, 128);
    // domain: [15000,64] @ [64,128]
    cvt_pack2_kernel<<<(15000 * 64 / 2 + 255) / 256, 256, 0, stream>>>(dom_feats, featb, 15000 * 64 / 2);
    pack_b_kernel<<<((64 / 32) * 128 * 16 + 255) / 256, 256, 0, stream>>>(Wd, Bpack, 64);
    wmma_gemm128_kernel<true><<<(15000 + 15) / 16, 256, 0, stream>>>(
        featb, 64, Bpack, bdv, xbuf, dom_idx, 15000, 64);
    // ip: [10000,32] @ [32,128]
    cvt_pack2_kernel<<<(10000 * 32 / 2 + 255) / 256, 256, 0, stream>>>(ip_feats, featb, 10000 * 32 / 2);
    pack_b_kernel<<<((32 / 32) * 128 * 16 + 255) / 256, 256, 0, stream>>>(Wi, Bpack, 32);
    wmma_gemm128_kernel<true><<<(10000 + 15) / 16, 256, 0, stream>>>(
        featb, 32, Bpack, biv, xbuf, ip_idx, 10000, 32);

    // ---- one RGCN layer
    auto layer = [&](const float* xin, const float* comp, const float* bases,
                     const float* root, const float* bias,
                     const float* lg, const float* lb,
                     float* ln_out, int relu) {
        zero_v4_kernel<<<(N * NREL * HID / 4 + 255) / 256, 256, 0, stream>>>(
            (float4*)msum, N * NREL * HID / 4);
        zero_v4_kernel<<<(N * NREL / 4 + 255) / 256, 256, 0, stream>>>(
            (float4*)cnt, N * NREL / 4);
        scatter_mean_kernel<<<(E + 7) / 8, 256, 0, stream>>>(
            edge_index, edge_type, xin, msum, cnt, E);
        pack_rgcn_b_kernel<<<((KEXT / 32) * 128 * 16 + 255) / 256, 256, 0, stream>>>(
            comp, bases, root, Bpack);
        meanpack_kernel<<<(N * (KEXT / 2) + 255) / 256, 256, 0, stream>>>(
            msum, cnt, xin, Apack, N);
        wmma_gemm128_kernel<false><<<N / 16, 256, 0, stream>>>(
            Apack, KEXT, Bpack, bias, hbuf, nullptr, N, KEXT);
        layernorm_kernel<<<(N + 7) / 8, 256, 0, stream>>>(hbuf, lg, lb, ln_out, N, relu);
    };

    // layer 1: LN+ReLU output overwrites xbuf as next-layer input
    layer(xbuf, comp1, bases1, root1, bias1, ln1g, ln1b, xbuf, 1);
    // layer 2: final LN straight to d_out
    layer(xbuf, comp2, bases2, root2, bias2, ln2g, ln2b, (float*)d_out, 0);
}
